// EncoderAttention_56401510531618
// MI455X (gfx1250) — matrix operand order repaired and checked by reading, not compiled, not run on hardware
//
#include <hip/hip_runtime.h>
#include <hip/hip_bf16.h>

// ---------------------------------------------------------------------------
// EncoderAttention for MI455X (gfx1250, wave32, WMMA f32_16x16x32_f16)
//   B=4, P=2048, D_MODEL=768, H=12, D_HEAD=64
// convert(+transpose) | QKV GEMM | flash attention | output GEMM
// - contiguous ds_load_b128 fragment reads; B fragments reused across 2 row
//   groups (8 WMMA per k-step per wave)
// - global->LDS staging = GLOBAL_LOAD_ASYNC_TO_LDS_B128 (ASYNCcnt), double
//   buffered, consecutive chunks via IOFFSET immediates
// - softmax scale folded into Q so flash uses bare v_exp_f32
// ---------------------------------------------------------------------------

typedef __attribute__((ext_vector_type(16))) _Float16 v16h;
typedef __attribute__((ext_vector_type(8)))  float    v8f;
typedef _Float16 half_t;

#define H_    12
#define DM    768
#define DH    64
#define B_    4
#define P_    2048
#define M_TOT (B_ * P_)          // 8192

// 1/sqrt(64) * log2(e): folded into Q so flash softmax uses exp2
#define QSCALE 0.18033688011112042f

// Async global->LDS b128 with compile-time IOFFSET (added to BOTH addresses).
#define CP_ASYNC_O(laddr, gptr, OFF)                                      \
  asm volatile("global_load_async_to_lds_b128 %0, %1, off offset:" #OFF   \
               :: "v"(laddr), "v"(gptr) : "memory")
#define ASYNC_WAIT(n) asm volatile("s_wait_asynccnt %0" :: "i"(n) : "memory")

static __device__ __forceinline__ unsigned lds_addr(const void* p) {
  return (unsigned)(uintptr_t)p;          // low 32 bits of flat LDS ptr = offset
}

static __device__ __forceinline__ v8f wmma_f16(v16h a, v16h b, v8f c) {
  return __builtin_amdgcn_wmma_f32_16x16x32_f16(
      false, a, false, b, (short)0, c, false, false);
}

// A fragment (16x32 f16) from row-major [16 x stride] tile: contiguous reads.
static __device__ __forceinline__ v16h fragA_16x32(const half_t* p, int stride) {
  const int lane = threadIdx.x & 31;
  const int row  = lane & 15;
  const int kb   = (lane >> 4) << 4;          // 0 or 16
  const half_t* r = p + row * stride;
  v16h a;
#pragma unroll
  for (int i = 0; i < 16; ++i) a[i] = r[kb + i];
  return a;
}

// B fragment (32x16) from an [N x K] row-major tile: element (k,n)=p[n*stride+k].
static __device__ __forceinline__ v16h fragB_row(const half_t* p, int stride) {
  const int lane = threadIdx.x & 31;
  const int col  = lane & 15;
  const int kb   = (lane >> 4) << 4;          // 0 or 16
  const half_t* r = p + col * stride + kb;
  v16h b;
#pragma unroll
  for (int i = 0; i < 16; ++i) b[i] = r[i];
  return b;
}

// ---------------------------------------------------------------------------
__global__ void f32_to_f16_kernel(const float* __restrict__ src,
                                  half_t* __restrict__ dst, int n) {
  int i = blockIdx.x * blockDim.x + threadIdx.x;
  int stride = gridDim.x * blockDim.x;
  for (; i < n; i += stride) dst[i] = (half_t)src[i];
}

// Per-block transpose: src [nblk][R][C] f32 -> dst [nblk][C][R] f16.
template <int R, int C>
__global__ void transpose_f32_to_f16_kernel(const float* __restrict__ src,
                                            half_t* __restrict__ dst, int nblk) {
  int i = blockIdx.x * blockDim.x + threadIdx.x;
  int total = nblk * R * C;
  int stride = gridDim.x * blockDim.x;
  for (; i < total; i += stride) {
    int blk = i / (R * C);
    int rem = i - blk * (R * C);
    int r = rem / C, c = rem - r * C;
    dst[(size_t)blk * (R * C) + (size_t)c * R + r] = (half_t)src[i];
  }
}

// ---------------------------------------------------------------------------
// QKV projection: [8192 x 768] x Wt[64 x 768] per (mat, head), fused bias.
// 256 threads (8 waves); BM=256 (32 rows/wave), BN=64, BK=32.
// grid = (32, 3=mat, 12=head) -- no integer division.
// Q pre-scaled by QSCALE. Q,K: [B][H][P][64]; V transposed: [B][H][64][P].
// ---------------------------------------------------------------------------
__global__ __launch_bounds__(256) void qkv_gemm_kernel(
    const half_t* __restrict__ xh, const half_t* __restrict__ wqkvT,
    const float* __restrict__ qb, const float* __restrict__ kbias,
    const float* __restrict__ vb, half_t* __restrict__ qkv) {
  __shared__ __align__(16) half_t At[2][256 * 32];   // 16 KB each
  __shared__ __align__(16) half_t Bt[2][64 * 32];    //  4 KB each
  const int tid  = threadIdx.x;
  const int lane = tid & 31;
  const int w    = tid >> 5;
  const int rowBase = blockIdx.x * 256;
  const int mat = blockIdx.y;                 // 0=Q 1=K 2=V
  const int h   = blockIdx.z;
  const half_t* Wt = wqkvT + (size_t)(mat * H_ + h) * (DM * DH); // [64][768]
  const float* bias = (mat == 0 ? qb : (mat == 1 ? kbias : vb)) + h * DH;

  // Staging: thread = one A row (4 chunks @ +0/16/32/48B); 1 B chunk.
  const int nB = tid >> 2, cB = (tid & 3) << 3;
  const half_t* aSrc = xh + (size_t)(rowBase + tid) * DM;
  const half_t* bSrc = Wt + (size_t)nB * DM + cB;
  const unsigned aL0 = lds_addr(&At[0][tid * 32]);
  const unsigned bL0 = lds_addr(&Bt[0][nB * 32 + cB]);

  auto stage = [&](int buf, int kt) {
    unsigned al = aL0 + buf * (256 * 32 * 2);
    unsigned bl = bL0 + buf * (64 * 32 * 2);
    const half_t* as = aSrc + kt * 32;
    const half_t* bs = bSrc + kt * 32;
    CP_ASYNC_O(al, as, 0);  CP_ASYNC_O(al, as, 16);
    CP_ASYNC_O(al, as, 32); CP_ASYNC_O(al, as, 48);
    CP_ASYNC_O(bl, bs, 0);
  };

  v8f acc0[4] = {}, acc1[4] = {};
  stage(0, 0);
  for (int kt = 0; kt < DM / 32; ++kt) {
    int cur = kt & 1;
    __syncthreads();                          // buffer cur^1 free to overwrite
    if (kt + 1 < DM / 32) { stage(cur ^ 1, kt + 1); ASYNC_WAIT(5); }
    else                  { ASYNC_WAIT(0); }
    __syncthreads();
    // Hoist all fragment loads, then run the 8-WMMA chain.
    v16h a0 = fragA_16x32(&At[cur][(w * 32) * 32], 32);
    v16h a1 = fragA_16x32(&At[cur][(w * 32 + 16) * 32], 32);
    v16h bf[4];
#pragma unroll
    for (int t = 0; t < 4; ++t) bf[t] = fragB_row(&Bt[cur][(t * 16) * 32], 32);
#pragma unroll
    for (int t = 0; t < 4; ++t) acc0[t] = wmma_f16(a0, bf[t], acc0[t]);
#pragma unroll
    for (int t = 0; t < 4; ++t) acc1[t] = wmma_f16(a1, bf[t], acc1[t]);
  }

  const int rb = (lane >> 4) << 3;
  const int cn = lane & 15;
#pragma unroll
  for (int g = 0; g < 2; ++g) {
    v8f* acc = g ? acc1 : acc0;
#pragma unroll
    for (int t = 0; t < 4; ++t) {
      int col = t * 16 + cn;
      float bv = bias[col];
#pragma unroll
      for (int r = 0; r < 8; ++r) {
        int row = rowBase + w * 32 + g * 16 + rb + r;
        int bb = row >> 11, pp = row & (P_ - 1);
        float val = acc[t][r] + bv;
        if (mat == 0) val *= QSCALE;          // fold softmax scale into Q
        if (mat == 2) {   // V transposed: [B][H][DH][P]
          qkv[(size_t)2 * M_TOT * DH +
              (((size_t)bb * H_ + h) * DH + col) * P_ + pp] = (half_t)val;
        } else {          // Q/K: [B][H][P][DH]
          qkv[(size_t)mat * M_TOT * DH +
              (((size_t)bb * H_ + h) * P_ + pp) * DH + col] = (half_t)val;
        }
      }
    }
  }
}

// ---------------------------------------------------------------------------
// Flash attention per (b,h): 128 threads (4 waves), 64 query rows per block.
// K staged [key][d]; V staged from transposed [d][key]; double-buffered async.
// ---------------------------------------------------------------------------
__global__ __launch_bounds__(128) void flash_attn_kernel(
    const half_t* __restrict__ qbuf, const half_t* __restrict__ kbuf,
    const half_t* __restrict__ vT, half_t* __restrict__ zbuf) {
  __shared__ __align__(16) half_t Kt[2][64 * 64];   // [key][d]
  __shared__ __align__(16) half_t Vt[2][64 * 64];   // [d][key]
  __shared__ __align__(16) half_t Pt[4 * 16 * 64];  // per-wave P transpose area
  const int tid  = threadIdx.x;
  const int lane = tid & 31;
  const int w    = tid >> 5;
  const int b    = blockIdx.z, h = blockIdx.y;
  const int qBase = blockIdx.x * 64;
  const size_t headOff = ((size_t)b * H_ + h) * (size_t)P_ * DH;

  // Staging: row rS = tid>>1 (0..63), 4 consecutive chunks per thread.
  const int rS = tid >> 1, cS = (tid & 1) << 5;     // 0 or 32 halfs
  const half_t* kSrc = kbuf + headOff + (size_t)rS * DH + cS;
  const half_t* vSrc = vT   + headOff + (size_t)rS * P_ + cS;
  const unsigned kL0 = lds_addr(&Kt[0][rS * 64 + cS]);
  const unsigned vL0 = lds_addr(&Vt[0][rS * 64 + cS]);

  auto stage = [&](int buf, int kb0) {
    unsigned kl = kL0 + buf * (64 * 64 * 2);
    unsigned vl = vL0 + buf * (64 * 64 * 2);
    const half_t* ks = kSrc + (size_t)kb0 * DH;
    const half_t* vs = vSrc + kb0;
    CP_ASYNC_O(kl, ks, 0);  CP_ASYNC_O(kl, ks, 16);
    CP_ASYNC_O(kl, ks, 32); CP_ASYNC_O(kl, ks, 48);
    CP_ASYNC_O(vl, vs, 0);  CP_ASYNC_O(vl, vs, 16);
    CP_ASYNC_O(vl, vs, 32); CP_ASYNC_O(vl, vs, 48);
  };

  // Q fragments (pre-scaled by QSCALE in qkv_gemm) live in registers.
  const half_t* Q = qbuf + headOff + (size_t)(qBase + w * 16) * DH;
  v16h qa0 = fragA_16x32(Q, DH);
  v16h qa1 = fragA_16x32(Q + 32, DH);

  v8f acc[4] = {};
  float mrow[8], lrow[8];
#pragma unroll
  for (int r = 0; r < 8; ++r) { mrow[r] = -3.0e38f; lrow[r] = 0.0f; }
  half_t* Pw = Pt + w * (16 * 64);
  const int rb = (lane >> 4) << 3;
  const int cn = lane & 15;

  stage(0, 0);
  for (int i = 0; i < P_ / 64; ++i) {
    int cur = i & 1;
    __syncthreads();                          // prev compute done -> cur^1 free
    if (i + 1 < P_ / 64) { stage(cur ^ 1, (i + 1) * 64); ASYNC_WAIT(8); }
    else                 { ASYNC_WAIT(0); }
    __syncthreads();

    // S = Q * K^T   (16 x 64 per wave; scores already in log2e units)
    v8f s[4];
#pragma unroll
    for (int t = 0; t < 4; ++t) {
      v8f z = {};
      z = wmma_f16(qa0, fragB_row(&Kt[cur][(t * 16) * DH], DH), z);
      z = wmma_f16(qa1, fragB_row(&Kt[cur][(t * 16) * DH] + 32, DH), z);
      s[t] = z;
    }

    // Online softmax (exp2-based). Rows rb..rb+7 live in this 16-lane group.
    float mnew[8], alpha[8];
#pragma unroll
    for (int r = 0; r < 8; ++r) {
      float v = fmaxf(fmaxf(s[0][r], s[1][r]), fmaxf(s[2][r], s[3][r]));
#pragma unroll
      for (int off = 1; off < 16; off <<= 1)
        v = fmaxf(v, __shfl_xor(v, off, 32));
      mnew[r]  = fmaxf(mrow[r], v);
      alpha[r] = exp2f(mrow[r] - mnew[r]);
      mrow[r]  = mnew[r];
    }
    float rsum[8];
#pragma unroll
    for (int r = 0; r < 8; ++r) rsum[r] = 0.0f;
#pragma unroll
    for (int t = 0; t < 4; ++t) {
#pragma unroll
      for (int r = 0; r < 8; ++r) {
        float p = exp2f(s[t][r] - mnew[r]);
        rsum[r] += p;
        Pw[(rb + r) * 64 + t * 16 + cn] = (half_t)p;  // C-layout -> LDS
      }
    }
#pragma unroll
    for (int r = 0; r < 8; ++r) {
      float v = rsum[r];
#pragma unroll
      for (int off = 1; off < 16; off <<= 1)
        v += __shfl_xor(v, off, 32);
      lrow[r] = lrow[r] * alpha[r] + v;
#pragma unroll
      for (int t = 0; t < 4; ++t) acc[t][r] = acc[t][r] * alpha[r];
    }
    __syncthreads();

    // O += P * V   (P as A-fragments; V^T tile gives contiguous B reads)
    v16h pa0 = fragA_16x32(Pw, 64);
    v16h pa1 = fragA_16x32(Pw + 32, 64);
#pragma unroll
    for (int t = 0; t < 4; ++t) {
      acc[t] = wmma_f16(pa0, fragB_row(&Vt[cur][(t * 16) * 64], 64), acc[t]);
      acc[t] = wmma_f16(pa1, fragB_row(&Vt[cur][(t * 16) * 64] + 32, 64), acc[t]);
    }
  }

  // z[b, q, h*64 + d]  (f16, laid out for the output GEMM)
#pragma unroll
  for (int r = 0; r < 8; ++r) {
    float rinv = 1.0f / lrow[r];
    int q = qBase + w * 16 + rb + r;
#pragma unroll
    for (int t = 0; t < 4; ++t)
      zbuf[((size_t)b * P_ + q) * DM + h * DH + t * 16 + cn] =
          (half_t)(acc[t][r] * rinv);
  }
}

// ---------------------------------------------------------------------------
// Output projection: z[8192 x 768] x O^T[768 out][768 in] + Ob -> out f32
// BM=256 (32 rows/wave), BN=64; B fragments shared across both row groups.
// ---------------------------------------------------------------------------
__global__ __launch_bounds__(256) void out_gemm_kernel(
    const half_t* __restrict__ z, const half_t* __restrict__ owT,
    const float* __restrict__ ob, float* __restrict__ out) {
  __shared__ __align__(16) half_t At[2][256 * 32];
  __shared__ __align__(16) half_t Bt[2][64 * 32];
  const int tid = threadIdx.x, lane = tid & 31, w = tid >> 5;
  const int rowBase = blockIdx.x * 256;
  const int colBase = blockIdx.y * 64;

  const int nB = tid >> 2, cB = (tid & 3) << 3;
  const half_t* aSrc = z   + (size_t)(rowBase + tid) * DM;
  const half_t* bSrc = owT + (size_t)(colBase + nB) * DM + cB;
  const unsigned aL0 = lds_addr(&At[0][tid * 32]);
  const unsigned bL0 = lds_addr(&Bt[0][nB * 32 + cB]);

  auto stage = [&](int buf, int kt) {
    unsigned al = aL0 + buf * (256 * 32 * 2);
    unsigned bl = bL0 + buf * (64 * 32 * 2);
    const half_t* as = aSrc + kt * 32;
    const half_t* bs = bSrc + kt * 32;
    CP_ASYNC_O(al, as, 0);  CP_ASYNC_O(al, as, 16);
    CP_ASYNC_O(al, as, 32); CP_ASYNC_O(al, as, 48);
    CP_ASYNC_O(bl, bs, 0);
  };

  v8f acc0[4] = {}, acc1[4] = {};
  stage(0, 0);
  for (int kt = 0; kt < DM / 32; ++kt) {
    int cur = kt & 1;
    __syncthreads();
    if (kt + 1 < DM / 32) { stage(cur ^ 1, kt + 1); ASYNC_WAIT(5); }
    else                  { ASYNC_WAIT(0); }
    __syncthreads();
    v16h a0 = fragA_16x32(&At[cur][(w * 32) * 32], 32);
    v16h a1 = fragA_16x32(&At[cur][(w * 32 + 16) * 32], 32);
    v16h bf[4];
#pragma unroll
    for (int t = 0; t < 4; ++t) bf[t] = fragB_row(&Bt[cur][(t * 16) * 32], 32);
#pragma unroll
    for (int t = 0; t < 4; ++t) acc0[t] = wmma_f16(a0, bf[t], acc0[t]);
#pragma unroll
    for (int t = 0; t < 4; ++t) acc1[t] = wmma_f16(a1, bf[t], acc1[t]);
  }

  const int rb = (lane >> 4) << 3;
  const int cn = lane & 15;
#pragma unroll
  for (int g = 0; g < 2; ++g) {
    v8f* acc = g ? acc1 : acc0;
#pragma unroll
    for (int t = 0; t < 4; ++t) {
      int col = colBase + t * 16 + cn;
      float bv = ob[col];
#pragma unroll
      for (int r = 0; r < 8; ++r) {
        int row = rowBase + w * 32 + g * 16 + rb + r;
        out[(size_t)row * DM + col] = acc[t][r] + bv;
      }
    }
  }
}

// ---------------------------------------------------------------------------
extern "C" void kernel_launch(void* const* d_in, const int* in_sizes, int n_in,
                              void* d_out, int out_size, void* d_ws, size_t ws_size,
                              hipStream_t stream) {
  const float* x   = (const float*)d_in[0];
  const float* Qs  = (const float*)d_in[1];
  const float* Ks  = (const float*)d_in[2];
  const float* Vs  = (const float*)d_in[3];
  const float* O   = (const float*)d_in[4];
  const float* Qbs = (const float*)d_in[5];
  const float* Kbs = (const float*)d_in[6];
  const float* Vbs = (const float*)d_in[7];
  const float* Ob  = (const float*)d_in[8];
  float* out = (float*)d_out;

  half_t* ws = (half_t*)d_ws;
  const size_t NX = (size_t)M_TOT * DM;        // 6,291,456
  const size_t NW = (size_t)H_ * DM * DH;      //   589,824
  half_t* xh    = ws;  ws += NX;
  half_t* wqkvT = ws;  ws += 3 * NW;           // [3][H][64][768] (transposed)
  half_t* owT   = ws;  ws += NW;               // [768 out][768 in] (transposed)
  half_t* qkv   = ws;  ws += 3 * NX;           // q | k | vT
  half_t* zb    = ws;  ws += NX;               // z [B][P][768]

  f32_to_f16_kernel<<<1024, 256, 0, stream>>>(x, xh, (int)NX);
  transpose_f32_to_f16_kernel<DM, DH><<<256, 256, 0, stream>>>(Qs, wqkvT,          H_);
  transpose_f32_to_f16_kernel<DM, DH><<<256, 256, 0, stream>>>(Ks, wqkvT + NW,     H_);
  transpose_f32_to_f16_kernel<DM, DH><<<256, 256, 0, stream>>>(Vs, wqkvT + 2 * NW, H_);
  transpose_f32_to_f16_kernel<DM, DM><<<256, 256, 0, stream>>>(O,  owT,            1);

  qkv_gemm_kernel<<<dim3(M_TOT / 256, 3, H_), 256, 0, stream>>>(
      xh, wqkvT, Qbs, Kbs, Vbs, qkv);
  flash_attn_kernel<<<dim3(P_ / 64, H_, B_), 128, 0, stream>>>(
      qkv, qkv + NX, qkv + 2 * NX, zb);
  out_gemm_kernel<<<dim3(M_TOT / 256, DM / 64), 256, 0, stream>>>(
      zb, owT, Ob, out);
}
